// EGAT_80771154968600
// MI455X (gfx1250) — matrix operand order repaired and checked by reading, not compiled, not run on hardware
//
#include <hip/hip_runtime.h>
#include <hip/hip_bf16.h>

typedef _Float16 half_t;
typedef __attribute__((ext_vector_type(16))) _Float16 v16h;
typedef __attribute__((ext_vector_type(8)))  _Float16 v8h;
typedef __attribute__((ext_vector_type(4)))  _Float16 v4h;
typedef __attribute__((ext_vector_type(8)))  float    v8f;

static constexpr int   NN        = 2048;
static constexpr int   PP        = 4;
static constexpr int   FH        = 64;
static constexpr int   HH        = 4;
static constexpr int   CC        = 16;
static constexpr int   FIN       = 256;
static constexpr int   FCAT      = HH * PP * FH;   // 1024
static constexpr int   ATT_STRIDE = NN + 8;        // pad to dodge LDS bank conflicts
static constexpr float ALPHA     = 0.2f;

__device__ __forceinline__ float lrelu(float x) { return x > 0.f ? x : ALPHA * x; }

// ---------------------------------------------------------------------------
// Kernel 1: Wh = X @ W  (WMMA f16 inputs, f32 accumulate). One wave per 16x16
// C tile. Writes f32 (for f_src/f_dst) and a TRANSPOSED f16 copy Wh16T[f][j]
// (leading dim N) so the einsum's B fragments become contiguous-K b128 loads.
// ---------------------------------------------------------------------------
__global__ __launch_bounds__(32) void gemm_wh_kernel(
    const float* __restrict__ X, const float* __restrict__ Wm,
    float* __restrict__ Wh32, half_t* __restrict__ Wh16T,
    int Fin, int Fout, int N) {
  const int lane  = threadIdx.x;
  const int i0    = blockIdx.x * 16;
  const int c0    = blockIdx.y * 16;
  const int Ml    = lane & 15;
  const bool hiL  = lane >= 16;
  const int abase = hiL ? 16 : 0;
  const int koff  = hiL ? 16 : 0;

  v8f C;
#pragma unroll
  for (int r = 0; r < 8; ++r) C[r] = 0.f;

  for (int k0 = 0; k0 < Fin; k0 += 32) {
    // A fragment: rows of X, f32 -> f16
    const float* xr = X + (size_t)(i0 + Ml) * Fin + k0 + abase;
    v16h A;
#pragma unroll
    for (int e = 0; e < 8; ++e) { A[e] = (half_t)xr[e]; A[8 + e] = (half_t)xr[8 + e]; }
    // B fragment: column of W (strided), f32 -> f16 (tiny GEMM, cost ok)
    const float* wr = Wm + (size_t)(k0 + koff) * Fout + c0 + Ml;
    v16h B;
#pragma unroll
    for (int e = 0; e < 16; ++e) B[e] = (half_t)wr[(size_t)e * Fout];
    C = __builtin_amdgcn_wmma_f32_16x16x32_f16(false, A, false, B, (short)0, C, false, false);
  }
  const int col = c0 + Ml;
#pragma unroll
  for (int r = 0; r < 8; ++r) {
    const int row = i0 + r + (hiL ? 8 : 0);
    Wh32[(size_t)row * Fout + col]  = C[r];
    Wh16T[(size_t)col * N + row]    = (half_t)C[r];   // contiguous along r
  }
}

// ---------------------------------------------------------------------------
// Kernel 2: f_src/f_dst = Wh @ a_src / a_dst
// ---------------------------------------------------------------------------
__global__ void fvec_kernel(const float* __restrict__ Wh,
                            const float* __restrict__ a_src,
                            const float* __restrict__ a_dst,
                            float* __restrict__ fsrc, float* __restrict__ fdst,
                            int F, int N) {
  const int i = blockIdx.x * blockDim.x + threadIdx.x;
  if (i >= N) return;
  float s0 = 0.f, s1 = 0.f;
  for (int f = 0; f < F; ++f) {
    const float w = Wh[(size_t)i * F + f];
    s0 += w * a_src[f];
    s1 += w * a_dst[f];
  }
  fsrc[i] = s0;
  fdst[i] = s1;
}

// ---------------------------------------------------------------------------
// Kernel 3: fused edge-GAT attention.
// One workgroup (16 waves) = 16 rows of one p-slice. Each wave keeps its full
// 2048-score row in 64 VGPRs/lane: score -> shfl max -> exp/sum -> normalize.
// att is written back (f32, next head's edge_attr) and staged in LDS (f16) as
// the WMMA A operand; h' = att @ Wh via v_wmma_f32_16x16x32_f16 with K split
// across waves and ds_add_f32 reduction. B fragments come from the transposed
// Wh16T: 2x b128 contiguous loads per fragment.
// ---------------------------------------------------------------------------
template <int FOUT, bool HEADS>
__global__ __launch_bounds__(512) void att_kernel(
    const float* eaSrc, float* eaDst,              // may alias (in-place rows)
    const half_t* __restrict__ Wh16T,              // [FOUT][N]
    const float* __restrict__ fsrc, const float* __restrict__ fdst,
    float* __restrict__ outp, int outStride, int colBase, int N) {
  extern __shared__ char smem_raw[];
  half_t* attS = (half_t*)smem_raw;                               // [16][ATT_STRIDE]
  float*  accS = (float*)(smem_raw + (size_t)16 * ATT_STRIDE * sizeof(half_t)); // [16][FOUT]

  const int lane = threadIdx.x & 31;
  const int wave = threadIdx.x >> 5;    // 0..15
  const int i0   = blockIdx.x * 16;
  const int p    = blockIdx.y;
  const int i    = i0 + wave;

  // ---------------- Phase 1: row softmax, row held in registers -------------
  const float  fi = fsrc[i];
  const float* ea = eaSrc + ((size_t)p * N + i) * N;
  float4 s[16];
  float  m = -INFINITY;
#pragma unroll
  for (int t = 0; t < 16; ++t) {
    const float4 v  = ((const float4*)ea)[t * 32 + lane];
    const float4 fd = ((const float4*)fdst)[t * 32 + lane];
    float4 sc;
    sc.x = lrelu(fi + fd.x) * v.x;
    sc.y = lrelu(fi + fd.y) * v.y;
    sc.z = lrelu(fi + fd.z) * v.z;
    sc.w = lrelu(fi + fd.w) * v.w;
    s[t] = sc;
    m = fmaxf(m, fmaxf(fmaxf(sc.x, sc.y), fmaxf(sc.z, sc.w)));
  }
#pragma unroll
  for (int o = 16; o > 0; o >>= 1) m = fmaxf(m, __shfl_xor(m, o, 32));

  float l = 0.f;
#pragma unroll
  for (int t = 0; t < 16; ++t) {
    s[t].x = __expf(s[t].x - m); s[t].y = __expf(s[t].y - m);
    s[t].z = __expf(s[t].z - m); s[t].w = __expf(s[t].w - m);
    l += s[t].x + s[t].y + s[t].z + s[t].w;
  }
#pragma unroll
  for (int o = 16; o > 0; o >>= 1) l += __shfl_xor(l, o, 32);
  const float inv = 1.0f / l;

  float4* eo = HEADS ? (float4*)(eaDst + ((size_t)p * N + i) * N) : nullptr;
#pragma unroll
  for (int t = 0; t < 16; ++t) {
    float4 a4;
    a4.x = s[t].x * inv; a4.y = s[t].y * inv;
    a4.z = s[t].z * inv; a4.w = s[t].w * inv;
    if (HEADS) eo[t * 32 + lane] = a4;     // new edge_attr for next layer
    v4h h4;
    h4[0] = (half_t)a4.x; h4[1] = (half_t)a4.y;
    h4[2] = (half_t)a4.z; h4[3] = (half_t)a4.w;
    *(v4h*)(attS + (size_t)wave * ATT_STRIDE + t * 128 + lane * 4) = h4;
  }
  __syncthreads();

  // ---------------- Phase 2: h' = att @ Wh via WMMA -------------------------
  for (int idx = threadIdx.x; idx < 16 * FOUT; idx += blockDim.x) accS[idx] = 0.f;
  __syncthreads();

  const int  Ml    = lane & 15;
  const bool hiL   = lane >= 16;
  const int  abase = hiL ? 16 : 0;
  const int  koff  = hiL ? 16 : 0;
  const int  kbase = wave * (N / 16);    // 128-wide K slice per wave
  constexpr int NFB = FOUT / 16;

  v8f Cacc[NFB];
#pragma unroll
  for (int fb = 0; fb < NFB; ++fb)
#pragma unroll
    for (int r = 0; r < 8; ++r) Cacc[fb][r] = 0.f;

  for (int ks = 0; ks < (N / 16) / 32; ++ks) {   // 4 K-steps of 32
    const int k0 = kbase + ks * 32;
    const half_t* ar = attS + (size_t)Ml * ATT_STRIDE + k0 + abase;
    const v8h alo = *(const v8h*)(ar);
    const v8h ahi = *(const v8h*)(ar + 8);
    v16h A;
#pragma unroll
    for (int e = 0; e < 8; ++e) { A[e] = alo[e]; A[8 + e] = ahi[e]; }
#pragma unroll
    for (int fb = 0; fb < NFB; ++fb) {
      // B fragment from transposed Wh: 16 contiguous K halves per lane
      const half_t* br = Wh16T + (size_t)(fb * 16 + Ml) * N + k0 + koff;
      const v8h b0 = *(const v8h*)(br);
      const v8h b1 = *(const v8h*)(br + 8);
      v16h B;
#pragma unroll
      for (int e = 0; e < 8; ++e) { B[e] = b0[e]; B[8 + e] = b1[e]; }
      Cacc[fb] = __builtin_amdgcn_wmma_f32_16x16x32_f16(false, A, false, B, (short)0,
                                                        Cacc[fb], false, false);
    }
  }
#pragma unroll
  for (int fb = 0; fb < NFB; ++fb)
#pragma unroll
    for (int r = 0; r < 8; ++r) {
      const int M = r + (hiL ? 8 : 0);
      atomicAdd(&accS[M * FOUT + fb * 16 + Ml], Cacc[fb][r]);
    }
  __syncthreads();

  // ---------------- Output ---------------------------------------------------
  for (int idx = threadIdx.x; idx < 16 * FOUT; idx += blockDim.x) {
    const int M = idx / FOUT, f = idx % FOUT;
    float v = accS[idx];
    if (HEADS) {  // ELU + concat layout: xcat[i, h*256 + p*64 + f]
      v = v > 0.f ? v : __expf(v) - 1.f;
      outp[(size_t)(i0 + M) * outStride + colBase + p * FOUT + f] = v;
    } else {      // h2[p, i, c]
      outp[((size_t)p * N + (i0 + M)) * FOUT + f] = v;
    }
  }
}

// ---------------------------------------------------------------------------
// Kernel 4: logits[i] = log_softmax(elu(mean_p h2[p,i,:]))
// ---------------------------------------------------------------------------
__global__ void logits_kernel(const float* __restrict__ h2, float* __restrict__ out, int N) {
  const int i = blockIdx.x * blockDim.x + threadIdx.x;
  if (i >= N) return;
  float v[CC];
#pragma unroll
  for (int c = 0; c < CC; ++c) {
    float a = 0.f;
#pragma unroll
    for (int p = 0; p < PP; ++p) a += h2[((size_t)p * N + i) * CC + c];
    a *= 0.25f;
    v[c] = a > 0.f ? a : __expf(a) - 1.f;   // elu
  }
  float m = -INFINITY;
#pragma unroll
  for (int c = 0; c < CC; ++c) m = fmaxf(m, v[c]);
  float l = 0.f;
#pragma unroll
  for (int c = 0; c < CC; ++c) l += __expf(v[c] - m);
  const float lg = __logf(l);
#pragma unroll
  for (int c = 0; c < CC; ++c) out[(size_t)i * CC + c] = v[c] - m - lg;
}

// ---------------------------------------------------------------------------
extern "C" void kernel_launch(void* const* d_in, const int* in_sizes, int n_in,
                              void* d_out, int out_size, void* d_ws, size_t ws_size,
                              hipStream_t stream) {
  (void)in_sizes; (void)n_in; (void)out_size; (void)ws_size;
  const float* x       = (const float*)d_in[0];  // [N, FIN]
  const float* ea_in   = (const float*)d_in[1];  // [P, N, N]
  const float* W_heads = (const float*)d_in[2];  // [H, FIN, FH]
  const float* a_src_h = (const float*)d_in[3];  // [H, FH]
  const float* a_dst_h = (const float*)d_in[4];  // [H, FH]
  const float* W_out   = (const float*)d_in[5];  // [FCAT, CC]
  const float* a_src_o = (const float*)d_in[6];  // [CC]
  const float* a_dst_o = (const float*)d_in[7];  // [CC]
  float* out = (float*)d_out;                    // [N, CC]

  // Workspace layout (~73 MB)
  float*  EA    = (float*)d_ws;                   // P*N*N
  float*  XCAT  = EA + (size_t)PP * NN * NN;      // N*FCAT
  float*  WH32  = XCAT + (size_t)NN * FCAT;       // N*FH (reused for final N*CC)
  half_t* WH16T = (half_t*)(WH32 + (size_t)NN * FH); // FH*N halves (transposed)
  float*  FS    = (float*)(WH16T + (size_t)NN * FH);
  float*  FD    = FS + NN;
  float*  H2    = FD + NN;                        // P*N*CC

  const size_t shHeads = (size_t)16 * ATT_STRIDE * sizeof(half_t) + (size_t)16 * FH * sizeof(float);
  const size_t shFinal = (size_t)16 * ATT_STRIDE * sizeof(half_t) + (size_t)16 * CC * sizeof(float);

  for (int h = 0; h < HH; ++h) {
    gemm_wh_kernel<<<dim3(NN / 16, FH / 16), 32, 0, stream>>>(
        x, W_heads + (size_t)h * FIN * FH, WH32, WH16T, FIN, FH, NN);
    fvec_kernel<<<(NN + 255) / 256, 256, 0, stream>>>(
        WH32, a_src_h + h * FH, a_dst_h + h * FH, FS, FD, FH, NN);
    const float* src = (h == 0) ? ea_in : EA;   // head 0 reads immutable input
    att_kernel<FH, true><<<dim3(NN / 16, PP), 512, shHeads, stream>>>(
        src, EA, WH16T, FS, FD, XCAT, FCAT, h * PP * FH, NN);
  }

  // Final layer: xcat @ W_out, then attention with C=16, mean over p.
  gemm_wh_kernel<<<dim3(NN / 16, CC / 16), 32, 0, stream>>>(
      XCAT, W_out, WH32, WH16T, FCAT, CC, NN);
  fvec_kernel<<<(NN + 255) / 256, 256, 0, stream>>>(
      WH32, a_src_o, a_dst_o, FS, FD, CC, NN);
  att_kernel<CC, false><<<dim3(NN / 16, PP), 512, shFinal, stream>>>(
      EA, nullptr, WH16T, FS, FD, H2, 0, 0, NN);
  logits_kernel<<<(NN + 255) / 256, 256, 0, stream>>>(H2, out, NN);
}
